// HeteroGAE_geo_Decoder_33818572489424
// MI455X (gfx1250) — matrix operand order repaired and checked by reading, not compiled, not run on hardware
//
#include <hip/hip_runtime.h>

// ---------------------------------------------------------------------------
// HeteroGAE geo decoder for MI455X (gfx1250, wave32).
//
// Roofline: ~520 GFLOP dense GEMM -> v_wmma_f32_16x16x32_bf16 (f32 accum);
// ~5 GB scattered/streamed activations -> atomic scatter-mean + fused
// epilogues. Plain-A GEMM tiles are staged LDS via the Tensor Data Mover
// (tensor_load_to_lds, TENSORcnt) with double buffering.
// ---------------------------------------------------------------------------

#define NN 200000
#define EE 800000
#define PP 500000
#define DD 128
#define HH 256

typedef __attribute__((ext_vector_type(16))) __bf16 v16bf;
typedef __attribute__((ext_vector_type(8)))  __bf16 v8bf;
typedef __attribute__((ext_vector_type(2)))  __bf16 v2bf;
typedef __attribute__((ext_vector_type(8)))  float  v8f;
typedef __attribute__((ext_vector_type(4)))  unsigned v4u;
typedef __attribute__((ext_vector_type(8)))  unsigned v8u;

__device__ __forceinline__ float gelu_f(float x) {
  return 0.5f * x * (1.0f + erff(x * 0.70710678118654752f));
}

// A-fragment (16x32 bf16) from a bf16 LDS tile, per CDNA5 A layout:
// lanes 0-15: M=lane,    elems 0..7 -> K 0..7,   elems 8..15 -> K 16..23
// lanes16-31: M=lane-16, elems 0..7 -> K 8..15,  elems 8..15 -> K 24..31
__device__ __forceinline__ v16bf lds_load_a(const __bf16* As, int lane) {
  int m = lane & 15;
  int koff = (lane < 16) ? 0 : 8;
  const __bf16* row = As + m * 32;
  v8bf lo = *(const v8bf*)(row + koff);
  v8bf hi = *(const v8bf*)(row + 16 + koff);
  v16bf a;
#pragma unroll
  for (int j = 0; j < 8; ++j) { a[j] = lo[j]; a[j + 8] = hi[j]; }
  return a;
}

// Same fragment sourced from an f32 LDS tile (TDM-staged), converting in-reg.
__device__ __forceinline__ v16bf lds_load_a_f32(const float* Af, int lane) {
  int m = lane & 15;
  int koff = (lane < 16) ? 0 : 8;
  const float* row = Af + m * 32;
  v16bf a;
#pragma unroll
  for (int j = 0; j < 8; ++j) {
    a[j]     = (__bf16)row[koff + j];
    a[j + 8] = (__bf16)row[16 + koff + j];
  }
  return a;
}

// B fragments pre-packed: elem j of lane l -> K=(j<8?0:16)+(l<16?0:8)+(j&7), N=l&15 (32B/lane).
__device__ __forceinline__ v16bf load_b_packed(const __bf16* Bp, int kt, int ntiles,
                                               int nt, int lane) {
  return *(const v16bf*)(Bp + (((long)(kt * ntiles + nt)) * 32 + lane) * 16);
}

// Issue a TDM load of a 16x32 f32 tile (row stride lda elems) into LDS.
// 2D descriptor: groups 2/3 omitted (NULL). Wave-uniform args required.
__device__ __forceinline__ void tdm_load_tile_f32(unsigned lds_off, const float* gptr,
                                                  int lda) {
  unsigned long ga = (unsigned long)gptr;
  v4u g0;
  g0[0] = 1u;                                               // count=1, user mode
  g0[1] = lds_off;                                          // lds_addr (bytes)
  g0[2] = (unsigned)ga;                                     // global_addr lo
  g0[3] = (unsigned)((ga >> 32) & 0x01FFFFFFu) | 0x80000000u; // addr hi | type=2
  v8u g1;
  g1[0] = 0x00020000u;       // workgroup_mask=0, data_size=4B, no pad/iterate
  g1[1] = 32u << 16;         // tensor_dim0 = 32 (lo16 in bits 63:48)
  g1[2] = 16u << 16;         // tensor_dim0 hi=0 | tensor_dim1 = 16
  g1[3] = 32u << 16;         // tensor_dim1 hi=0 | tile_dim0 = 32
  g1[4] = 16u;               // tile_dim1 = 16, tile_dim2 = 0
  g1[5] = (unsigned)lda;     // tensor_dim0_stride lo32 (elements)
  g1[6] = 0u;                // stride hi | tensor_dim1_stride lo
  g1[7] = 0u;
  asm volatile("tensor_load_to_lds %0, %1" :: "s"(g0), "s"(g1) : "memory");
}

// ---------------------------------------------------------------------------
__global__ void zero_f(float* p, long n) {
  long i = (long)blockIdx.x * 256 + threadIdx.x;
  if (i < n) p[i] = 0.0f;
}

// Pack a row-major f32 K x Nout weight into WMMA-B bf16 lane layout.
__global__ void pack_b(const float* W, __bf16* Bp, int K, int Nout) {
  long e = (long)blockIdx.x * 256 + threadIdx.x;
  long total = (long)K * Nout;
  if (e >= total) return;
  int ntiles = Nout >> 4;
  long per_kt = (long)ntiles * 512;
  int kt = (int)(e / per_kt);
  long rem = e % per_kt;
  int nt = (int)(rem / 512);
  int rem2 = (int)(rem % 512);
  int lane = rem2 >> 4, j = rem2 & 15;
  int k = kt * 32 + ((j & 8) << 1) + ((lane >> 4) << 3) + (j & 7);
  int n = nt * 16 + (lane & 15);
  Bp[e] = (__bf16)W[(long)k * Nout + n];
}

// Scatter-add of source-node features into msum[dst], plus degree count.
__global__ void scatter_k(const float* x, const int* src, const int* dst,
                          float* msum, float* deg, int E, int din) {
  int chunks = din >> 2;
  long tid = (long)blockIdx.x * 256 + threadIdx.x;
  if (tid >= (long)E * chunks) return;
  int e = (int)(tid / chunks), c = (int)(tid % chunks);
  int s = src[e], d = dst[e];
  float4 v = *(const float4*)(x + (long)s * din + c * 4);
  float* mp = msum + (long)d * din + c * 4;
  atomicAdd(mp + 0, v.x);
  atomicAdd(mp + 1, v.y);
  atomicAdd(mp + 2, v.z);
  atomicAdd(mp + 3, v.w);
  if (c == 0) atomicAdd(deg + d, 1.0f);
}

// SAGE layer GEMM: hpre = gelu(mean_msg @ Wl + x @ Wr + b), bf16 WMMA.
// Block = 8 waves on one 16-row M strip; Nout=256 (2 N-tiles/wave).
__global__ void sage_gemm(const float* msum, const float* deg, const float* xin, int din,
                          const __bf16* BpL, const __bf16* BpR, const float* bias,
                          float* Out) {
  __shared__ __attribute__((aligned(16))) __bf16 Am[16 * 32];
  __shared__ __attribute__((aligned(16))) __bf16 Ax[16 * 32];
  __shared__ float rdeg[16];
  int tid = threadIdx.x, wave = tid >> 5, lane = tid & 31;
  int mbase = blockIdx.x * 16;
  if (tid < 16) rdeg[tid] = 1.0f / fmaxf(deg[mbase + tid], 1.0f);
  __syncthreads();
  v8f acc[2];
#pragma unroll
  for (int t = 0; t < 2; ++t)
#pragma unroll
    for (int j = 0; j < 8; ++j) acc[t][j] = 0.0f;
  int nkt = din >> 5;
  for (int kt = 0; kt < nkt; ++kt) {
    {  // stage both A tiles: 2 consecutive K elems per thread (float2 loads)
      int e = tid << 1;
      int r = e >> 5, c = e & 31;
      long row = mbase + r;
      int k = (kt << 5) + c;
      float2 vm = *(const float2*)(msum + row * din + k);
      float2 vx = *(const float2*)(xin + row * din + k);
      float rd = rdeg[r];
      v2bf pm; pm[0] = (__bf16)(vm.x * rd); pm[1] = (__bf16)(vm.y * rd);
      v2bf px; px[0] = (__bf16)vx.x;        px[1] = (__bf16)vx.y;
      *(v2bf*)(Am + e) = pm;
      *(v2bf*)(Ax + e) = px;
    }
    __syncthreads();
    v16bf am = lds_load_a(Am, lane);
    v16bf ax = lds_load_a(Ax, lane);
#pragma unroll
    for (int t = 0; t < 2; ++t) {
      int nt = wave + (t << 3);
      v16bf bl = load_b_packed(BpL, kt, 16, nt, lane);
      acc[t] = __builtin_amdgcn_wmma_f32_16x16x32_bf16(false, am, false, bl,
                                                       (short)0, acc[t], false, false);
      v16bf br = load_b_packed(BpR, kt, 16, nt, lane);
      acc[t] = __builtin_amdgcn_wmma_f32_16x16x32_bf16(false, ax, false, br,
                                                       (short)0, acc[t], false, false);
    }
    __syncthreads();
  }
  int rowoff = (lane >> 4) * 8, nlo = lane & 15;
#pragma unroll
  for (int t = 0; t < 2; ++t) {
    int col = (wave + (t << 3)) * 16 + nlo;
    float bv = bias[col];
#pragma unroll
    for (int r = 0; r < 8; ++r) {
      long row = mbase + rowoff + r;
      Out[row * 256 + col] = gelu_f(acc[t][r] + bv);
    }
  }
}

// Column sums + sums of squares (GraphNorm stats), h is M x 256.
__global__ void colstats(const float* h, int M, float* sum, float* sumsq) {
  int t = threadIdx.x;  // t == column
  int r0 = blockIdx.x * 64;
  int r1 = r0 + 64; if (r1 > M) r1 = M;
  float s = 0.0f, q = 0.0f;
  for (int r = r0; r < r1; ++r) {
    float v = h[(long)r * 256 + t];
    s += v; q += v * v;
  }
  atomicAdd(&sum[t], s);
  atomicAdd(&sumsq[t], q);
}

// mu = E[h]; var = E[h^2] - 2*ms*mu^2 + ms^2*mu^2; inv = rsqrt(var+eps)
__global__ void fin_stats(float* stats, const float* gns, float Mf) {
  int c = threadIdx.x;
  float mu = stats[c] / Mf;
  float e2 = stats[256 + c] / Mf;
  float ms = gns[c];
  float var = e2 - 2.0f * ms * mu * mu + ms * ms * mu * mu;
  stats[512 + c] = mu;
  stats[768 + c] = rsqrtf(var + 1e-5f);
}

// y = (h - ms*mu)*inv*w + b (+ prev if addPrev); write new h and z slice.
__global__ void gn_apply(const float* hpre, const float* stats, const float* gw,
                         const float* gb, const float* gns, float* h, float* zsl,
                         int addPrev) {
  long i = (long)blockIdx.x * 256 + threadIdx.x;
  int c = (int)(i & 255);
  float mu = stats[512 + c], inv = stats[768 + c];
  float y = (hpre[i] - gns[c] * mu) * inv * gw[c] + gb[c];
  if (addPrev) y += h[i];
  h[i] = y;
  zsl[(i >> 8) * 768 + c] = y;
}

// WMMA GEMM with transformed/gathered A staged to LDS as bf16.
// AM: 1 = DynamicTanh(A), 2 = gathered contact-pair A.  Epilogue: GELU.
template <int AM>
__global__ void gemm16(const float* A, int lda, const __bf16* Bp, const float* bias,
                       float* Out, int ldo, int K, int Nout,
                       const float* dalpha, const float* dytw, const float* dytb,
                       const int* pidx, int Pn, const float* zfp) {
  __shared__ __attribute__((aligned(16))) __bf16 As[16 * 32];
  int tid = threadIdx.x, wave = tid >> 5, lane = tid & 31;
  int mbase = blockIdx.x * 16;
  int ntiles = Nout >> 4;
  int perWave = ntiles >> 3;  // 1, 2 or 4
  float alpha = 0.0f;
  if (AM == 1) alpha = dalpha[0];
  v8f acc[4];
#pragma unroll
  for (int t = 0; t < 4; ++t)
#pragma unroll
    for (int j = 0; j < 8; ++j) acc[t][j] = 0.0f;
  int nkt = K >> 5;
  for (int kt = 0; kt < nkt; ++kt) {
    {  // 2 consecutive K elems per thread
      int e = tid << 1;
      int r = e >> 5, c = e & 31;
      long row = mbase + r;
      int k = (kt << 5) + c;
      float v0, v1;
      if (AM == 1) {
        float2 av = *(const float2*)(A + row * lda + k);
        float2 w = *(const float2*)(dytw + k);
        float2 b = *(const float2*)(dytb + k);
        v0 = tanhf(alpha * av.x) * w.x + b.x;
        v1 = tanhf(alpha * av.y) * w.y + b.y;
      } else {  // pair gather; k even, so k and k+1 use the same endpoint
        int idx = (k < 128) ? pidx[row] : pidx[(long)Pn + row];
        float2 av = *(const float2*)(zfp + (long)idx * 128 + (k & 127));
        v0 = av.x; v1 = av.y;
      }
      v2bf p; p[0] = (__bf16)v0; p[1] = (__bf16)v1;
      *(v2bf*)(As + e) = p;
    }
    __syncthreads();
    v16bf a = lds_load_a(As, lane);
    for (int t = 0; t < perWave; ++t) {
      int nt = wave + (t << 3);
      v16bf b = load_b_packed(Bp, kt, ntiles, nt, lane);
      acc[t] = __builtin_amdgcn_wmma_f32_16x16x32_bf16(false, a, false, b,
                                                       (short)0, acc[t], false, false);
    }
    __syncthreads();
  }
  int rowoff = (lane >> 4) * 8, nlo = lane & 15;
  for (int t = 0; t < perWave; ++t) {
    int col = (wave + (t << 3)) * 16 + nlo;
    float bv = bias[col];
#pragma unroll
    for (int r = 0; r < 8; ++r) {
      long row = mbase + rowoff + r;
      Out[row * ldo + col] = gelu_f(acc[t][r] + bv);
    }
  }
}

// WMMA GEMM with plain row-major f32 A staged to LDS by the Tensor Data
// Mover (double-buffered; wave 0 drives TDM, workgroup barrier releases).
// EM: 0 = GELU epilogue, 1 = no act + residual add.
template <int EM>
__global__ void gemm16_tdm(const float* A, int lda, const __bf16* Bp, const float* bias,
                           float* Out, int ldo, int K, int Nout,
                           const float* resid, int residCols) {
  // Only __shared__ allocation in this kernel -> LDS offsets 0 and 2048.
  __shared__ __attribute__((aligned(16))) float Af[2][16 * 32];
  int tid = threadIdx.x, wave = tid >> 5, lane = tid & 31;
  int mbase = blockIdx.x * 16;
  int ntiles = Nout >> 4;
  int perWave = ntiles >> 3;
  v8f acc[4];
#pragma unroll
  for (int t = 0; t < 4; ++t)
#pragma unroll
    for (int j = 0; j < 8; ++j) acc[t][j] = 0.0f;
  int nkt = K >> 5;
  const float* Abase = A + (long)mbase * lda;
  if (wave == 0) tdm_load_tile_f32(0u, Abase, lda);
  for (int kt = 0; kt < nkt; ++kt) {
    if (wave == 0) {
      if (kt + 1 < nkt) {
        tdm_load_tile_f32(((kt + 1) & 1) ? 2048u : 0u, Abase + (kt + 1) * 32, lda);
        __builtin_amdgcn_s_wait_tensorcnt(1);  // tile kt complete (in-order TDM)
      } else {
        __builtin_amdgcn_s_wait_tensorcnt(0);
      }
    }
    __syncthreads();
    v16bf a = lds_load_a_f32(Af[kt & 1], lane);
    for (int t = 0; t < perWave; ++t) {
      int nt = wave + (t << 3);
      v16bf b = load_b_packed(Bp, kt, ntiles, nt, lane);
      acc[t] = __builtin_amdgcn_wmma_f32_16x16x32_bf16(false, a, false, b,
                                                       (short)0, acc[t], false, false);
    }
    __syncthreads();
  }
  int rowoff = (lane >> 4) * 8, nlo = lane & 15;
  for (int t = 0; t < perWave; ++t) {
    int col = (wave + (t << 3)) * 16 + nlo;
    float bv = bias[col];
#pragma unroll
    for (int r = 0; r < 8; ++r) {
      long row = mbase + rowoff + r;
      float v = acc[t][r] + bv;
      if (EM == 0) v = gelu_f(v);
      else v += resid[row * residCols + col];
      Out[row * ldo + col] = v;
    }
  }
}

// Row-wise L2 normalization of zf (M x 128); one wave per row.
__global__ void rownorm(float* zf, int M) {
  int row = blockIdx.x * 8 + (threadIdx.x >> 5);
  int lane = threadIdx.x & 31;
  if (row >= M) return;
  float4 v = *(const float4*)(zf + (long)row * 128 + lane * 4);
  float s = v.x * v.x + v.y * v.y + v.z * v.z + v.w * v.w;
#pragma unroll
  for (int off = 16; off; off >>= 1) s += __shfl_xor(s, off, 32);
  float r = 1.0f / (sqrtf(s) + 1e-10f);
  v.x *= r; v.y *= r; v.z *= r; v.w *= r;
  *(float4*)(zf + (long)row * 128 + lane * 4) = v;
}

// scores = sigmoid(e2 @ C3 + c3)
__global__ void score_k(const float* e2, const float* C3, const float* c3,
                        float* out, int P) {
  int p = blockIdx.x * 256 + threadIdx.x;
  if (p >= P) return;
  const float* row = e2 + (long)p * 128;
  float s = c3[0];
#pragma unroll
  for (int k = 0; k < 128; k += 4) {
    float4 v = *(const float4*)(row + k);
    float4 w = *(const float4*)(C3 + k);
    s += v.x * w.x + v.y * w.y + v.z * w.z + v.w * w.w;
  }
  out[p] = 1.0f / (1.0f + expf(-s));
}

// ---------------------------------------------------------------------------
extern "C" void kernel_launch(void* const* d_in, const int* in_sizes, int n_in,
                              void* d_out, int out_size, void* d_ws, size_t ws_size,
                              hipStream_t stream) {
  (void)in_sizes; (void)n_in; (void)out_size; (void)ws_size;
  const float* x_res = (const float*)d_in[0];
  const int*   eidx  = (const int*)d_in[1];
  const int*   cidx  = (const int*)d_in[2];
  const float* Wl[3]  = {(const float*)d_in[3],  (const float*)d_in[9],  (const float*)d_in[15]};
  const float* Wr[3]  = {(const float*)d_in[4],  (const float*)d_in[10], (const float*)d_in[16]};
  const float* bc[3]  = {(const float*)d_in[5],  (const float*)d_in[11], (const float*)d_in[17]};
  const float* gnw[3] = {(const float*)d_in[6],  (const float*)d_in[12], (const float*)d_in[18]};
  const float* gnb[3] = {(const float*)d_in[7],  (const float*)d_in[13], (const float*)d_in[19]};
  const float* gns[3] = {(const float*)d_in[8],  (const float*)d_in[14], (const float*)d_in[20]};
  const float* dalpha = (const float*)d_in[21];
  const float* dytw   = (const float*)d_in[22];
  const float* dytb   = (const float*)d_in[23];
  const float* W1 = (const float*)d_in[24]; const float* b1 = (const float*)d_in[25];
  const float* W2 = (const float*)d_in[26]; const float* b2 = (const float*)d_in[27];
  const float* W3 = (const float*)d_in[28]; const float* b3 = (const float*)d_in[29];
  const float* C1 = (const float*)d_in[30]; const float* c1 = (const float*)d_in[31];
  const float* C2 = (const float*)d_in[32]; const float* c2 = (const float*)d_in[33];
  const float* C3 = (const float*)d_in[34]; const float* c3 = (const float*)d_in[35];
  float* out = (float*)d_out;

  // ---- workspace layout (256B-aligned slices) ----
  char* ws = (char*)d_ws;
  size_t off = 0;
  auto take = [&](size_t bytes) {
    size_t r = off;
    off += (bytes + 255) & ~(size_t)255;
    return r;
  };
  __bf16* WB   = (__bf16*)(ws + take(1146880ull * 2));   // packed bf16 weights
  float*  deg  = (float*)(ws + take((size_t)NN * 4));
  float*  stats= (float*)(ws + take(1024 * 4));          // sum|sumsq|mu|inv
  float*  h    = (float*)(ws + take((size_t)NN * HH * 4));
  float*  msum = (float*)(ws + take((size_t)NN * HH * 4));
  float*  hpre = (float*)(ws + take((size_t)NN * HH * 4));
  float*  z    = (float*)(ws + take((size_t)NN * 768 * 4));
  float* t1 = msum;   // N x 512 (spans msum+hpre, both multiples of 256B)
  float* t2 = z;      // N x 512 (z is free once t1 is built)
  float* zf = h;      // N x 128 (h is free after layer 3)
  float* e1 = z;      // P x 256 (t2 is consumed before e1 is written)
  float* e2 = msum;   // P x 128 (t1 span free after t2)

  // ---- pack all GEMM weights to bf16 WMMA-B layout (deterministic) ----
  struct PackInfo { const float* W; size_t o; int K; int Nout; };
  PackInfo packs[11] = {
      {Wl[0], 0,       128, 256}, {Wr[0], 32768,  128, 256},
      {Wl[1], 65536,   256, 256}, {Wr[1], 131072, 256, 256},
      {Wl[2], 196608,  256, 256}, {Wr[2], 262144, 256, 256},
      {W1,    327680,  768, 512}, {W2,    720896, 512, 512},
      {W3,    983040,  512, 128}, {C1,   1048576, 256, 256},
      {C2,   1114112,  256, 128}};
  for (int i = 0; i < 11; ++i) {
    long tot = (long)packs[i].K * packs[i].Nout;
    pack_b<<<dim3((unsigned)((tot + 255) / 256)), 256, 0, stream>>>(
        packs[i].W, WB + packs[i].o, packs[i].K, packs[i].Nout);
  }
  size_t wl_off[3] = {0, 65536, 196608};
  size_t wr_off[3] = {32768, 131072, 262144};

  const int* src = eidx;
  const int* dst = eidx + EE;

  // ---- 3 SAGE + GraphNorm layers ----
  for (int i = 0; i < 3; ++i) {
    int din = (i == 0) ? DD : HH;
    const float* xin = (i == 0) ? x_res : h;
    long msz = (long)NN * din;
    zero_f<<<dim3((unsigned)((msz + 255) / 256)), 256, 0, stream>>>(msum, msz);
    zero_f<<<dim3((NN + 255) / 256), 256, 0, stream>>>(deg, NN);
    zero_f<<<dim3(2), 256, 0, stream>>>(stats, 512);
    long sthr = (long)EE * (din >> 2);
    scatter_k<<<dim3((unsigned)((sthr + 255) / 256)), 256, 0, stream>>>(
        xin, src, dst, msum, deg, EE, din);
    sage_gemm<<<dim3(NN / 16), 256, 0, stream>>>(
        msum, deg, xin, din, WB + wl_off[i], WB + wr_off[i], bc[i], hpre);
    colstats<<<dim3((NN + 63) / 64), 256, 0, stream>>>(hpre, NN, stats, stats + 256);
    fin_stats<<<dim3(1), 256, 0, stream>>>(stats, gns[i], (float)NN);
    gn_apply<<<dim3((unsigned)(((long)NN * HH) / 256)), 256, 0, stream>>>(
        hpre, stats, gnw[i], gnb[i], gns[i], h, z + i * HH, (i > 0) ? 1 : 0);
  }

  // ---- DynamicTanh + node MLP (768 -> 512 -> 512 -> 128) + residual ----
  gemm16<1><<<dim3(NN / 16), 256, 0, stream>>>(
      z, 768, WB + 327680, b1, t1, 512, 768, 512,
      dalpha, dytw, dytb, nullptr, 0, nullptr);
  gemm16_tdm<0><<<dim3(NN / 16), 256, 0, stream>>>(
      t1, 512, WB + 720896, b2, t2, 512, 512, 512, nullptr, 0);
  gemm16_tdm<1><<<dim3(NN / 16), 256, 0, stream>>>(
      t2, 512, WB + 983040, b3, zf, 128, 512, 128, x_res, 128);
  rownorm<<<dim3(NN / 8), 256, 0, stream>>>(zf, NN);

  // ---- contact-pair scorer (gathered 256 -> 256 -> 128 -> 1) ----
  gemm16<2><<<dim3(PP / 16), 256, 0, stream>>>(
      nullptr, 0, WB + 1048576, c1, e1, 256, 256, 256,
      nullptr, nullptr, nullptr, cidx, PP, zf);
  gemm16_tdm<0><<<dim3(PP / 16), 256, 0, stream>>>(
      e1, 256, WB + 1114112, c2, e2, 128, 256, 128, nullptr, 0);
  score_k<<<dim3((PP + 255) / 256), 256, 0, stream>>>(e2, C3, c3, out, PP);
}
